// GNNModel_29910152249929
// MI455X (gfx1250) — matrix operand order repaired and checked by reading, not compiled, not run on hardware
//
#include <hip/hip_runtime.h>
#include <hip/hip_bf16.h>
#include <math.h>

#define NN     102400
#define LL     16
#define DD     64
#define EE     819200
#define NPOOL  16384
#define BBATCH 64
#define SUBSG  256
#define EPOOL  (EE + NPOOL)
#define NEGV   (-1.0e9f)

typedef __attribute__((ext_vector_type(16))) _Float16 v16h;
typedef __attribute__((ext_vector_type(8)))  _Float16 v8h;
typedef __attribute__((ext_vector_type(8)))  float    v8f;

// ---------------- operand prep ----------------
__global__ void cvt_f16(const float* __restrict__ in, _Float16* __restrict__ out, long n) {
  long i  = (long)blockIdx.x * blockDim.x + threadIdx.x;
  long st = (long)gridDim.x * blockDim.x;
  for (; i < n; i += st) out[i] = (_Float16)in[i];
}

// Pack W[K,Nc] (f32 row-major) into WMMA-B fragment-major f16 blobs:
// blob(tn16, kb32) = 512 halves; within blob, lane l owns halves [l*16 .. l*16+15],
// element e -> W[kb32*32 + (l>=16?16:0) + e][tn16*16 + (l&15)].
// A lane's whole fragment is then one contiguous 32-byte read.
__global__ void pack_w(const float* __restrict__ W, _Float16* __restrict__ wp,
                       int K, int Nc) {
  int idx = blockIdx.x * 256 + threadIdx.x;
  if (idx >= K * Nc) return;
  int blob = idx >> 9, rem = idx & 511;
  int lane = rem >> 4, e = rem & 15;
  int kblocks = K >> 5;
  int tn16 = blob / kblocks, kb32 = blob % kblocks;
  int k   = (kb32 << 5) + ((lane & 16) ? 16 : 0) + e;
  int col = (tn16 << 4) + (lane & 15);
  wp[idx] = (_Float16)W[(size_t)k * Nc + col];
}

// ---------------- WMMA GEMM: C = act(A0@W0 (+A1@W1) + bias) ----------------
// Block: 256 thr = 8 waves -> 64(M) x 64(N) macro-tile; wave (wm,wn) owns 16 x 32
// (2 accumulators, A fragment reused). A staged via LDS (b128 in, b128 out);
// B fragments read as contiguous 32B from pre-packed weights.
__global__ void __launch_bounds__(256)
wmma_gemm2(const _Float16* __restrict__ A0, const _Float16* __restrict__ Wp0, int K0,
           const _Float16* __restrict__ A1, const _Float16* __restrict__ Wp1, int K1,
           const float* __restrict__ bias, float* __restrict__ C,
           int M, int Nc, int doRelu) {
  __shared__ _Float16 lA[64 * 32];
  const int tid = threadIdx.x;
  const int lane = tid & 31, wid = tid >> 5;
  const int wm = wid >> 1, wn = wid & 1;
  const int rowBase = blockIdx.y * 64;
  const int colBase = blockIdx.x * 64 + wn * 32;
  const bool act0 = (colBase < Nc);
  const bool act1 = (colBase + 16 < Nc);
  const int stR = tid >> 2, stS = tid & 3;            // staging: row, 8-half segment
  const int fRow = wm * 16 + (lane & 15);             // A fragment row in LDS tile
  const int fOff = ((lane >> 4) ? 16 : 0);            // +16 K for lanes 16..31
  v8f acc0 = {}, acc1 = {};

  const int kblocks0 = K0 >> 5;
  for (int kb = 0; kb < K0; kb += 32) {
    __syncthreads();
    {
      const uint4* g = (const uint4*)(A0 + (size_t)(rowBase + stR) * K0 + kb + stS * 8);
      __builtin_prefetch(A0 + (size_t)(rowBase + stR) * K0 + kb + 32, 0, 1);
      *((uint4*)(lA + stR * 32 + stS * 8)) = *g;      // ds_store_b128
    }
    __syncthreads();
    const v8h* pa = (const v8h*)(lA + (size_t)fRow * 32 + fOff);
    v8h a0 = pa[0], a1 = pa[1];                        // 2x ds_load_b128
    v16h a = __builtin_shufflevector(a0, a1, 0,1,2,3,4,5,6,7,8,9,10,11,12,13,14,15);
    if (act0) {
      const v8h* pb = (const v8h*)(Wp0 + ((size_t)((colBase >> 4) * kblocks0 + (kb >> 5)) << 9)
                                        + lane * 16);
      v16h b = __builtin_shufflevector(pb[0], pb[1], 0,1,2,3,4,5,6,7,8,9,10,11,12,13,14,15);
      acc0 = __builtin_amdgcn_wmma_f32_16x16x32_f16(false, a, false, b,
                                                    (short)0, acc0, false, false);
    }
    if (act1) {
      const v8h* pb = (const v8h*)(Wp0 + ((size_t)(((colBase >> 4) + 1) * kblocks0 + (kb >> 5)) << 9)
                                        + lane * 16);
      v16h b = __builtin_shufflevector(pb[0], pb[1], 0,1,2,3,4,5,6,7,8,9,10,11,12,13,14,15);
      acc1 = __builtin_amdgcn_wmma_f32_16x16x32_f16(false, a, false, b,
                                                    (short)0, acc1, false, false);
    }
  }
  if (A1) {
    const int kblocks1 = K1 >> 5;
    for (int kb = 0; kb < K1; kb += 32) {
      __syncthreads();
      {
        const uint4* g = (const uint4*)(A1 + (size_t)(rowBase + stR) * K1 + kb + stS * 8);
        __builtin_prefetch(A1 + (size_t)(rowBase + stR) * K1 + kb + 32, 0, 1);
        *((uint4*)(lA + stR * 32 + stS * 8)) = *g;
      }
      __syncthreads();
      const v8h* pa = (const v8h*)(lA + (size_t)fRow * 32 + fOff);
      v8h a0 = pa[0], a1 = pa[1];
      v16h a = __builtin_shufflevector(a0, a1, 0,1,2,3,4,5,6,7,8,9,10,11,12,13,14,15);
      if (act0) {
        const v8h* pb = (const v8h*)(Wp1 + ((size_t)((colBase >> 4) * kblocks1 + (kb >> 5)) << 9)
                                          + lane * 16);
        v16h b = __builtin_shufflevector(pb[0], pb[1], 0,1,2,3,4,5,6,7,8,9,10,11,12,13,14,15);
        acc0 = __builtin_amdgcn_wmma_f32_16x16x32_f16(false, a, false, b,
                                                      (short)0, acc0, false, false);
      }
      if (act1) {
        const v8h* pb = (const v8h*)(Wp1 + ((size_t)(((colBase >> 4) + 1) * kblocks1 + (kb >> 5)) << 9)
                                          + lane * 16);
        v16h b = __builtin_shufflevector(pb[0], pb[1], 0,1,2,3,4,5,6,7,8,9,10,11,12,13,14,15);
        acc1 = __builtin_amdgcn_wmma_f32_16x16x32_f16(false, a, false, b,
                                                      (short)0, acc1, false, false);
      }
    }
  }
  // C/D layout: VGPR r -> M = r + (lane>=16 ? 8 : 0); N = lane&15
  const int rb = rowBase + wm * 16 + ((lane >> 4) << 3);
  for (int t = 0; t < 2; ++t) {
    if (t == 0 ? !act0 : !act1) continue;
    const v8f* ac = (t == 0) ? &acc0 : &acc1;
    int col = colBase + t * 16 + (lane & 15);
    float bv = bias ? bias[col] : 0.0f;
#pragma unroll
    for (int r = 0; r < 8; ++r) {
      float v = (*ac)[r] + bv;
      if (doRelu) v = fmaxf(v, 0.0f);
      C[(size_t)(rb + r) * Nc + col] = v;
    }
  }
}

// ---------------- elementwise / scatter kernels ----------------
__global__ void fill_f32(float* __restrict__ p, float v, long n) {
  long i  = (long)blockIdx.x * blockDim.x + threadIdx.x;
  long st = (long)gridDim.x * blockDim.x;
  for (; i < n; i += st) p[i] = v;
}

__global__ void embed_mean(const int* __restrict__ x, const float* __restrict__ emb,
                           float* __restrict__ h) {
  int n = blockIdx.x, d = threadIdx.x;
  const int* xr = x + (size_t)n * LL;
  float s = 0.f, c = 0.f;
#pragma unroll
  for (int l = 0; l < LL; ++l) {
    int t = xr[l];
    if (t != 0) { s += emb[(size_t)t * DD + d]; c += 1.f; }
  }
  h[(size_t)n * DD + d] = s / fmaxf(c, 1.f);
}

__global__ void edge_scatter(const int* __restrict__ src, const int* __restrict__ dst,
                             const float* __restrict__ h, float* __restrict__ nsum,
                             float* __restrict__ ncnt) {
  long idx = (long)blockIdx.x * blockDim.x + threadIdx.x;
  if (idx >= (long)EE * DD) return;
  int e = (int)(idx >> 6), d = (int)(idx & 63);
  int s = src[e], t = dst[e];
  atomicAdd(&nsum[(size_t)t * DD + d], h[(size_t)s * DD + d]);
  if (d == 0) atomicAdd(&ncnt[t], 1.f);
}

__global__ void seg_div(float* __restrict__ s, const float* __restrict__ c, long rows) {
  long idx = (long)blockIdx.x * blockDim.x + threadIdx.x;
  if (idx >= rows * DD) return;
  s[idx] = s[idx] / fmaxf(c[idx >> 6], 1.f);
}

__global__ void pool_scatter(const int* __restrict__ pidx, const float* __restrict__ h,
                             float* __restrict__ ps, float* __restrict__ pc) {
  long idx = (long)blockIdx.x * blockDim.x + threadIdx.x;
  if (idx >= (long)NN * DD) return;
  int n = (int)(idx >> 6), d = (int)(idx & 63);
  int p = pidx[n];
  atomicAdd(&ps[(size_t)p * DD + d], h[idx]);
  if (d == 0) atomicAdd(&pc[p], 1.f);
}

__global__ void build_pedges(const int* __restrict__ src, const int* __restrict__ dst,
                             const int* __restrict__ pidx, int* __restrict__ sp,
                             int* __restrict__ dp) {
  int e = blockIdx.x * 256 + threadIdx.x;
  if (e >= EPOOL) return;
  if (e < EE) {
    int s = pidx[src[e]], t = pidx[dst[e]];
    sp[e] = (s == t) ? -1 : s;   // drop pooled self-edges
    dp[e] = t;
  } else {
    int p = e - EE;              // explicit self-loops added by GATConv
    sp[e] = p; dp[e] = p;
  }
}

__global__ void attn_coeff(const float* __restrict__ xw, const float* __restrict__ asrc,
                           const float* __restrict__ adst, float* __restrict__ es,
                           float* __restrict__ ed, int H) {
  int idx = blockIdx.x * 256 + threadIdx.x;
  if (idx >= NPOOL * H) return;
  int n = idx / H, hh = idx % H;
  const float* xr = xw + (size_t)n * H * DD + (size_t)hh * DD;
  const float* as = asrc + (size_t)hh * DD;
  const float* ad = adst + (size_t)hh * DD;
  float s0 = 0.f, s1 = 0.f;
  for (int c = 0; c < DD; ++c) { float v = xr[c]; s0 += v * as[c]; s1 += v * ad[c]; }
  es[idx] = s0; ed[idx] = s1;
}

__device__ __forceinline__ void atomicMaxF(float* addr, float val) {
  if (val >= 0.f) atomicMax((int*)addr, __float_as_int(val));
  else            atomicMin((unsigned int*)addr, __float_as_uint(val));
}

__global__ void attn_e(const int* __restrict__ sp, const int* __restrict__ dp,
                       const float* __restrict__ es, const float* __restrict__ ed,
                       float* __restrict__ ebuf, float* __restrict__ m, int H) {
  long idx = (long)blockIdx.x * blockDim.x + threadIdx.x;
  if (idx >= (long)EPOOL * H) return;
  int e = (int)(idx / H), hh = (int)(idx % H);
  int s = sp[e], t = dp[e];
  float v = NEGV;
  if (s >= 0) {
    float q = es[s * H + hh] + ed[t * H + hh];
    v = (q > 0.f) ? q : 0.2f * q;            // leaky_relu 0.2
    atomicMaxF(&m[t * H + hh], v);
  }
  ebuf[idx] = v;
}

__global__ void attn_p(const int* __restrict__ dp, float* __restrict__ ebuf,
                       const float* __restrict__ m, float* __restrict__ dnm, int H) {
  long idx = (long)blockIdx.x * blockDim.x + threadIdx.x;
  if (idx >= (long)EPOOL * H) return;
  int e = (int)(idx / H), hh = (int)(idx % H);
  float p = __expf(ebuf[idx] - m[dp[e] * H + hh]);  // invalid -> exp(-1e9-m) == 0
  ebuf[idx] = p;
  atomicAdd(&dnm[dp[e] * H + hh], p);
}

__global__ void attn_agg(const int* __restrict__ sp, const int* __restrict__ dp,
                         const float* __restrict__ ebuf, const float* __restrict__ dnm,
                         const float* __restrict__ xw, float* __restrict__ outacc, int H) {
  long idx = (long)blockIdx.x * blockDim.x + threadIdx.x;
  if (idx >= (long)EPOOL * H) return;
  int e = (int)(idx / H), hh = (int)(idx % H);
  int s = sp[e];
  if (s < 0) return;
  int t = dp[e];
  float alpha = ebuf[idx] / fmaxf(dnm[t * H + hh], 1e-16f);
  const float* xr = xw + (size_t)s * H * DD + (size_t)hh * DD;
  float* orow = outacc + (size_t)t * H * DD + (size_t)hh * DD;
  for (int c = 0; c < DD; ++c) atomicAdd(&orow[c], alpha * xr[c]);
}

__global__ void bias_relu(const float* __restrict__ in, const float* __restrict__ b,
                          float* __restrict__ out, long rows, int cols) {
  long idx = (long)blockIdx.x * blockDim.x + threadIdx.x;
  if (idx >= rows * cols) return;
  out[idx] = fmaxf(in[idx] + b[idx % cols], 0.f);
}

__global__ void graph_scatter(const float* __restrict__ hp, float* __restrict__ g) {
  long idx = (long)blockIdx.x * blockDim.x + threadIdx.x;
  if (idx >= (long)NPOOL * DD) return;
  int p = (int)(idx >> 6), d = (int)(idx & 63);
  atomicAdd(&g[(size_t)(p / SUBSG) * DD + d], hp[idx]);  // batch_pooled[p] == p/SUBS
}

__global__ void scale_f32(float* __restrict__ p, float s, long n) {
  long i = (long)blockIdx.x * blockDim.x + threadIdx.x;
  if (i < n) p[i] *= s;
}

// ---------------- host orchestration ----------------
static inline int gblk(long n) {
  long b = (n + 255) / 256;
  return (int)(b > 262144 ? 262144 : b);
}

static void gemm(const float* A0, const float* W0, int K0,
                 const float* A1, const float* W1, int K1,
                 const float* bias, float* C, int M, int Nc, int relu,
                 _Float16* Ah0, _Float16* Ah1, _Float16* Wp0, _Float16* Wp1,
                 hipStream_t st) {
  cvt_f16<<<gblk((long)M * K0), 256, 0, st>>>(A0, Ah0, (long)M * K0);
  pack_w<<<gblk((long)K0 * Nc), 256, 0, st>>>(W0, Wp0, K0, Nc);
  if (A1) {
    cvt_f16<<<gblk((long)M * K1), 256, 0, st>>>(A1, Ah1, (long)M * K1);
    pack_w<<<gblk((long)K1 * Nc), 256, 0, st>>>(W1, Wp1, K1, Nc);
  }
  dim3 grid((Nc + 63) / 64, M / 64);
  wmma_gemm2<<<grid, 256, 0, st>>>(Ah0, Wp0, K0, A1 ? Ah1 : nullptr, Wp1, K1,
                                   bias, C, M, Nc, relu);
}

extern "C" void kernel_launch(void* const* d_in, const int* in_sizes, int n_in,
                              void* d_out, int out_size, void* d_ws, size_t ws_size,
                              hipStream_t stream) {
  (void)in_sizes; (void)n_in; (void)out_size; (void)ws_size;
  const float* embed     = (const float*)d_in[0];
  const float* sageWl[2] = {(const float*)d_in[1], (const float*)d_in[2]};
  const float* sageBl[2] = {(const float*)d_in[3], (const float*)d_in[4]};
  const float* sageWr[2] = {(const float*)d_in[5], (const float*)d_in[6]};
  const float* gatW[3]   = {(const float*)d_in[7],  (const float*)d_in[8],  (const float*)d_in[9]};
  const float* asrc[3]   = {(const float*)d_in[10], (const float*)d_in[11], (const float*)d_in[12]};
  const float* adst[3]   = {(const float*)d_in[13], (const float*)d_in[14], (const float*)d_in[15]};
  const float* gatB[3]   = {(const float*)d_in[16], (const float*)d_in[17], (const float*)d_in[18]};
  const float* fc1W = (const float*)d_in[19]; const float* fc1b = (const float*)d_in[20];
  const float* fc2W = (const float*)d_in[21]; const float* fc2b = (const float*)d_in[22];
  const float* outW = (const float*)d_in[23]; const float* outb = (const float*)d_in[24];
  const int* x    = (const int*)d_in[25];
  const int* ei   = (const int*)d_in[26];
  const int* src  = ei;
  const int* dst  = ei + EE;
  const int* pidx = (const int*)d_in[27];
  float* yout = (float*)d_out;

  char* ws = (char*)d_ws;
  size_t off = 0;
  auto alloc = [&](size_t bytes) -> void* {
    void* p = ws + off;
    off = (off + bytes + 255) & ~(size_t)255;
    return p;
  };
  float* bufA = (float*)alloc((size_t)NN * DD * 4);      // h ping
  float* bufB = (float*)alloc((size_t)NN * DD * 4);      // h pong / hp
  float* bufC = (float*)alloc((size_t)NN * DD * 4);      // neigh sums / pooled / hp
  float* bufD = (float*)alloc((size_t)NPOOL * 256 * 4);  // GAT out accumulator
  float* ncnt = (float*)alloc((size_t)NN * 4);
  float* esb  = (float*)alloc((size_t)NPOOL * 4 * 4);
  float* edb  = (float*)alloc((size_t)NPOOL * 4 * 4);
  float* mb   = (float*)alloc((size_t)NPOOL * 4 * 4);
  float* dnm  = (float*)alloc((size_t)NPOOL * 4 * 4);
  float* ebuf = (float*)alloc((size_t)EPOOL * 4 * 4);
  int*   spb  = (int*)alloc((size_t)EPOOL * 4);
  int*   dpb  = (int*)alloc((size_t)EPOOL * 4);
  float* gbuf = (float*)alloc((size_t)BBATCH * DD * 4);
  float* z1   = (float*)alloc((size_t)BBATCH * DD * 4);
  float* z2   = (float*)alloc((size_t)BBATCH * DD * 4);
  _Float16* Ah0 = (_Float16*)alloc((size_t)NN * DD * 2); // f16 activations
  _Float16* Ah1 = (_Float16*)alloc((size_t)NN * DD * 2);
  _Float16* Wp0 = (_Float16*)alloc(65536);               // packed weights
  _Float16* Wp1 = (_Float16*)alloc(65536);

  // ---- node embedding: masked mean over tokens ----
  embed_mean<<<NN, DD, 0, stream>>>(x, embed, bufA);

  // ---- two SAGEConv layers: neigh mean scatter + fused dual WMMA GEMM ----
  float* hin  = bufA;
  float* hout = bufB;
  for (int l = 0; l < 2; ++l) {
    fill_f32<<<gblk((long)NN * DD), 256, 0, stream>>>(bufC, 0.f, (long)NN * DD);
    fill_f32<<<gblk(NN), 256, 0, stream>>>(ncnt, 0.f, NN);
    edge_scatter<<<gblk((long)EE * DD), 256, 0, stream>>>(src, dst, hin, bufC, ncnt);
    seg_div<<<gblk((long)NN * DD), 256, 0, stream>>>(bufC, ncnt, NN);
    gemm(bufC, sageWl[l], DD, hin, sageWr[l], DD, sageBl[l], hout, NN, DD, 1,
         Ah0, Ah1, Wp0, Wp1, stream);
    float* t = hin; hin = hout; hout = t;
  }
  // hin == bufA holds final node features

  // ---- pool to sub-graphs (segment mean) into bufC [NP, 64] ----
  fill_f32<<<gblk((long)NPOOL * DD), 256, 0, stream>>>(bufC, 0.f, (long)NPOOL * DD);
  fill_f32<<<gblk(NPOOL), 256, 0, stream>>>(ncnt, 0.f, NPOOL);
  pool_scatter<<<gblk((long)NN * DD), 256, 0, stream>>>(pidx, bufA, bufC, ncnt);
  seg_div<<<gblk((long)NPOOL * DD), 256, 0, stream>>>(bufC, ncnt, NPOOL);

  // ---- pooled edges (+self loops) ----
  build_pedges<<<(EPOOL + 255) / 256, 256, 0, stream>>>(src, dst, pidx, spb, dpb);

  // ---- three GATConv layers ----
  int Hh[3]   = {4, 1, 1};
  int Cin[3]  = {DD, 256, DD};
  int Cout[3] = {256, DD, DD};
  float* hpin[3]  = {bufC, bufB, bufC};
  float* hpout[3] = {bufB, bufC, bufB};
  for (int l = 0; l < 3; ++l) {
    int H = Hh[l];
    gemm(hpin[l], gatW[l], Cin[l], nullptr, nullptr, 0, nullptr, bufA, NPOOL, Cout[l], 0,
         Ah0, Ah1, Wp0, Wp1, stream);
    attn_coeff<<<(NPOOL * H + 255) / 256, 256, 0, stream>>>(bufA, asrc[l], adst[l], esb, edb, H);
    fill_f32<<<gblk((long)NPOOL * H), 256, 0, stream>>>(mb, NEGV, (long)NPOOL * H);
    fill_f32<<<gblk((long)NPOOL * H), 256, 0, stream>>>(dnm, 0.f, (long)NPOOL * H);
    fill_f32<<<gblk((long)NPOOL * Cout[l]), 256, 0, stream>>>(bufD, 0.f, (long)NPOOL * Cout[l]);
    attn_e<<<gblk((long)EPOOL * H), 256, 0, stream>>>(spb, dpb, esb, edb, ebuf, mb, H);
    attn_p<<<gblk((long)EPOOL * H), 256, 0, stream>>>(dpb, ebuf, mb, dnm, H);
    attn_agg<<<gblk((long)EPOOL * H), 256, 0, stream>>>(spb, dpb, ebuf, dnm, bufA, bufD, H);
    bias_relu<<<gblk((long)NPOOL * Cout[l]), 256, 0, stream>>>(bufD, gatB[l], hpout[l],
                                                              NPOOL, Cout[l]);
  }
  // hpout[2] == bufB holds [NP, 64]

  // ---- global mean pool + classifier MLP (WMMA) ----
  fill_f32<<<gblk((long)BBATCH * DD), 256, 0, stream>>>(gbuf, 0.f, (long)BBATCH * DD);
  graph_scatter<<<gblk((long)NPOOL * DD), 256, 0, stream>>>(bufB, gbuf);
  scale_f32<<<gblk((long)BBATCH * DD), 256, 0, stream>>>(gbuf, 1.0f / SUBSG, (long)BBATCH * DD);
  gemm(gbuf, fc1W, DD, nullptr, nullptr, 0, fc1b, z1, BBATCH, DD, 1,
       Ah0, Ah1, Wp0, Wp1, stream);
  gemm(z1,   fc2W, DD, nullptr, nullptr, 0, fc2b, z2, BBATCH, DD, 1,
       Ah0, Ah1, Wp0, Wp1, stream);
  gemm(z2,   outW, DD, nullptr, nullptr, 0, outb, yout, BBATCH, 32, 0,
       Ah0, Ah1, Wp0, Wp1, stream);
}
